// GraphAttention_28226525069528
// MI455X (gfx1250) — matrix operand order repaired and checked by reading, not compiled, not run on hardware
//
#include <hip/hip_runtime.h>
#include <hip/hip_bf16.h>
#include <stddef.h>

typedef __attribute__((ext_vector_type(16))) __bf16 v16bf;
typedef __attribute__((ext_vector_type(8)))  __bf16 v8bf;
typedef __attribute__((ext_vector_type(8)))  float  v8f;

#define NSEQ 2048
#define DMODEL 128
#define NB 16
#define MROW 2049  // mask row stride (N+1)

__device__ __forceinline__ v16bf join16(v8bf lo, v8bf hi) {
  return __builtin_shufflevector(lo, hi, 0,1,2,3,4,5,6,7,8,9,10,11,12,13,14,15);
}

__device__ __forceinline__ v8f wmma_bf16(v16bf a, v16bf b, v8f c) {
  // D = A(16x32 bf16) x B(32x16 bf16) + C(16x16 f32)
  return __builtin_amdgcn_wmma_f32_16x16x32_bf16(false, a, false, b, (short)0, c, false, false);
}

__device__ __forceinline__ v8f zero8() { v8f z = {0.f,0.f,0.f,0.f,0.f,0.f,0.f,0.f}; return z; }

// ---- DPP16 half-wave (16-lane) butterfly reductions: pure VALU, no DS ------
template<int CTRL>
__device__ __forceinline__ float dpp_mov(float x) {
  int i = __builtin_bit_cast(int, x);
  i = __builtin_amdgcn_update_dpp(0, i, CTRL, 0xF, 0xF, true);
  return __builtin_bit_cast(float, i);
}
// quad_perm(1,0,3,2)=0xB1, quad_perm(2,3,0,1)=0x4E, row_half_mirror=0x141, row_mirror=0x140
__device__ __forceinline__ float red16_max(float v) {
  v = fmaxf(v, dpp_mov<0xB1>(v));
  v = fmaxf(v, dpp_mov<0x4E>(v));
  v = fmaxf(v, dpp_mov<0x141>(v));
  v = fmaxf(v, dpp_mov<0x140>(v));
  return v;
}
__device__ __forceinline__ float red16_sum(float v) {
  v += dpp_mov<0xB1>(v);
  v += dpp_mov<0x4E>(v);
  v += dpp_mov<0x141>(v);
  v += dpp_mov<0x140>(v);
  return v;
}

// ---------------------------------------------------------------------------
// Generic  C[rows x 128] = A[rows x 128] @ W[128 x 128] + bias   (contract K=128)
// ---------------------------------------------------------------------------
template<bool A_BF16, bool OUT_F32>
__global__ __launch_bounds__(128) void gemm_d128(const void* __restrict__ Av,
                                                 const float* __restrict__ W,
                                                 const float* __restrict__ bias,
                                                 void* __restrict__ Cv, int rows)
{
  __shared__ __bf16 ldsWT[DMODEL * 136];
  const int tid = threadIdx.x;

  for (int i = tid; i < DMODEL * DMODEL / 4; i += 128) {
    const float4 f = ((const float4*)W)[i];
    const int k = i >> 5;
    const int c = (i & 31) << 2;
    ldsWT[(c + 0) * 136 + k] = (__bf16)f.x;
    ldsWT[(c + 1) * 136 + k] = (__bf16)f.y;
    ldsWT[(c + 2) * 136 + k] = (__bf16)f.z;
    ldsWT[(c + 3) * 136 + k] = (__bf16)f.w;
  }
  __syncthreads();

  const int wave = tid >> 5, lane = tid & 31;
  const int lane16 = lane & 15, hf = lane >> 4;
  const int tile = blockIdx.x * 4 + wave;
  const int m_base = (tile >> 3) * 16;
  const int n_tile = tile & 7;
  if (m_base >= rows) return;
  const int row = m_base + lane16;
  const int col = n_tile * 16 + lane16;
  const int hiK8 = hf * 8, hiK16 = hf * 16;

  v8f acc = zero8();
  #pragma unroll
  for (int kk = 0; kk < DMODEL; kk += 32) {
    v16bf afrag;
    if (A_BF16) {
      const __bf16* A = (const __bf16*)Av + (size_t)row * DMODEL;
      v8bf lo = *(const v8bf*)(A + kk + hiK8);
      v8bf hi = *(const v8bf*)(A + kk + 16 + hiK8);
      afrag = join16(lo, hi);
    } else {
      const float* A = (const float*)Av + (size_t)row * DMODEL;
      const float4 f0 = *(const float4*)(A + kk + hiK8);
      const float4 f1 = *(const float4*)(A + kk + hiK8 + 4);
      const float4 f2 = *(const float4*)(A + kk + 16 + hiK8);
      const float4 f3 = *(const float4*)(A + kk + 16 + hiK8 + 4);
      afrag[0]=(__bf16)f0.x; afrag[1]=(__bf16)f0.y; afrag[2]=(__bf16)f0.z; afrag[3]=(__bf16)f0.w;
      afrag[4]=(__bf16)f1.x; afrag[5]=(__bf16)f1.y; afrag[6]=(__bf16)f1.z; afrag[7]=(__bf16)f1.w;
      afrag[8]=(__bf16)f2.x; afrag[9]=(__bf16)f2.y; afrag[10]=(__bf16)f2.z; afrag[11]=(__bf16)f2.w;
      afrag[12]=(__bf16)f3.x; afrag[13]=(__bf16)f3.y; afrag[14]=(__bf16)f3.z; afrag[15]=(__bf16)f3.w;
    }
    const v8bf b0 = *(const v8bf*)&ldsWT[col * 136 + kk + hiK8];
    const v8bf b1 = *(const v8bf*)&ldsWT[col * 136 + kk + 16 + hiK8];
    acc = wmma_bf16(afrag, join16(b0, b1), acc);
  }

  const float bn = bias[col];
  #pragma unroll
  for (int r = 0; r < 8; ++r) {
    const int m = m_base + r + hf * 8;   // C layout: M = r + 8*half
    const float val = acc[r] + bn;
    if (OUT_F32) ((float*)Cv)[(size_t)m * DMODEL + col] = val;
    else         ((__bf16*)Cv)[(size_t)m * DMODEL + col] = (__bf16)val;
  }
}

// ---------------------------------------------------------------------------
// Flash attention: per wave one 16-row query tile; 4 waves/block share 64-key
// K/V LDS tiles. Online softmax with DPP reductions; P reshaped via LDS.
// ---------------------------------------------------------------------------
#define KT 64  // keys per iteration

__global__ __launch_bounds__(128) void attn_flash(const __bf16* __restrict__ Q,
                                                  const __bf16* __restrict__ K,
                                                  const __bf16* __restrict__ V,
                                                  const unsigned char* __restrict__ mask,
                                                  __bf16* __restrict__ O)
{
  __shared__ __bf16 ldsK[KT * 136];       // row-major: [key][d], pitch 136
  __shared__ __bf16 ldsVT[DMODEL * 72];   // transposed: [d][key], pitch 72
  __shared__ __bf16 ldsP[4][16 * KT];     // per-wave P tile (16 rows x 64 keys)

  const int tid = threadIdx.x, wave = tid >> 5, lane = tid & 31;
  const int lane16 = lane & 15, hf = lane >> 4;
  const int batch = blockIdx.y;
  const int m_base = (blockIdx.x * 4 + wave) * 16;

  const __bf16* Qb = Q + (size_t)batch * NSEQ * DMODEL;
  const __bf16* Kb = K + (size_t)batch * NSEQ * DMODEL;
  const __bf16* Vb = V + (size_t)batch * NSEQ * DMODEL;
  const unsigned char* Mb = mask + (size_t)batch * MROW * MROW;

  // preload the wave's Q fragments (A layout) for all 4 K-chunks
  v16bf qf[4];
  {
    const __bf16* qrow = Qb + (size_t)(m_base + lane16) * DMODEL;
    #pragma unroll
    for (int i = 0; i < 4; ++i) {
      const int kk = i * 32;
      v8bf lo = *(const v8bf*)(qrow + kk + hf * 8);
      v8bf hi = *(const v8bf*)(qrow + kk + 16 + hf * 8);
      qf[i] = join16(lo, hi);
    }
  }

  v8f oacc[8];
  #pragma unroll
  for (int d = 0; d < 8; ++d) oacc[d] = zero8();
  float Mr[8], Lr[8];
  const float NEG = -__builtin_inff();
  #pragma unroll
  for (int r = 0; r < 8; ++r) { Mr[r] = NEG; Lr[r] = 0.f; }

  const float scale = 0.08838834764831845f;  // 1/sqrt(128)

  for (int kb = 0; kb < NSEQ; kb += KT) {
    __syncthreads();
    // stage K (row-major) and V (transposed): 64 keys x 128 d, bf16
    for (int i = tid; i < KT * 16; i += 128) {
      const int r = i >> 4;            // key row 0..63
      const int c = (i & 15) << 3;     // d chunk start
      *(uint4*)&ldsK[r * 136 + c] = *(const uint4*)(Kb + (size_t)(kb + r) * DMODEL + c);
      uint4 vv = *(const uint4*)(Vb + (size_t)(kb + r) * DMODEL + c);
      const __bf16* vh = (const __bf16*)&vv;
      #pragma unroll
      for (int j = 0; j < 8; ++j) ldsVT[(c + j) * 72 + r] = vh[j];
    }
    __syncthreads();

    // scores: S[t] -> keys kb+16t .. kb+16t+15
    v8f S[4];
    #pragma unroll
    for (int t = 0; t < 4; ++t) S[t] = zero8();
    #pragma unroll
    for (int i = 0; i < 4; ++i) {
      const int kk = i * 32;
      #pragma unroll
      for (int t = 0; t < 4; ++t) {
        v8bf b0 = *(const v8bf*)&ldsK[(t * 16 + lane16) * 136 + kk + hf * 8];
        v8bf b1 = *(const v8bf*)&ldsK[(t * 16 + lane16) * 136 + kk + 16 + hf * 8];
        S[t] = wmma_bf16(qf[i], join16(b0, b1), S[t]);
      }
    }

    // mask + online softmax (row stats via DPP within each 16-lane half)
    #pragma unroll
    for (int r = 0; r < 8; ++r) {
      const int mrow = m_base + r + hf * 8 + 1;               // mask[:,1:,1:]
      const unsigned char* mrp = Mb + (size_t)mrow * MROW + 1 + kb;
      float s[4];
      #pragma unroll
      for (int t = 0; t < 4; ++t)
        s[t] = mrp[t * 16 + lane16] ? NEG : S[t][r] * scale;

      float v = fmaxf(fmaxf(s[0], s[1]), fmaxf(s[2], s[3]));
      v = red16_max(v);
      const float Mnew = fmaxf(Mr[r], v);

      float alpha, p[4];
      if (Mnew == NEG) {
        alpha = 1.f; p[0] = p[1] = p[2] = p[3] = 0.f;          // fully masked so far
      } else {
        alpha = __expf(Mr[r] - Mnew);                          // exp(-inf)=0 ok
        #pragma unroll
        for (int t = 0; t < 4; ++t)
          p[t] = (s[t] == NEG) ? 0.f : __expf(s[t] - Mnew);
      }
      float tsum = (p[0] + p[1]) + (p[2] + p[3]);
      tsum = red16_sum(tsum);
      Lr[r] = Lr[r] * alpha + tsum;
      Mr[r] = Mnew;
      #pragma unroll
      for (int d = 0; d < 8; ++d) oacc[d][r] *= alpha;

      __bf16* prow = &ldsP[wave][(r + hf * 8) * KT];
      prow[lane16]      = (__bf16)p[0];
      prow[16 + lane16] = (__bf16)p[1];
      prow[32 + lane16] = (__bf16)p[2];
      prow[48 + lane16] = (__bf16)p[3];
    }
    asm volatile("s_wait_dscnt 0" ::: "memory");  // wave-local C->A reshape via LDS

    // P (16x64, A layout, two K=32 chunks) @ V tile
    const __bf16* prow = &ldsP[wave][lane16 * KT];
    v16bf pf0 = join16(*(const v8bf*)(prow + hf * 8),
                       *(const v8bf*)(prow + 16 + hf * 8));
    v16bf pf1 = join16(*(const v8bf*)(prow + 32 + hf * 8),
                       *(const v8bf*)(prow + 48 + hf * 8));
    #pragma unroll
    for (int d = 0; d < 8; ++d) {
      const __bf16* vcol = &ldsVT[(d * 16 + lane16) * 72];
      v16bf vf0 = join16(*(const v8bf*)(vcol + hf * 8),
                         *(const v8bf*)(vcol + 16 + hf * 8));
      v16bf vf1 = join16(*(const v8bf*)(vcol + 32 + hf * 8),
                         *(const v8bf*)(vcol + 48 + hf * 8));
      oacc[d] = wmma_bf16(pf0, vf0, oacc[d]);
      oacc[d] = wmma_bf16(pf1, vf1, oacc[d]);
    }
  }

  // epilogue: divide by row sums (L==0 -> 0, matching NaN->0 semantics)
  #pragma unroll
  for (int r = 0; r < 8; ++r) {
    const int m = m_base + r + hf * 8;
    const float L = Lr[r];
    const float inv = (L > 0.f) ? 1.f / L : 0.f;
    #pragma unroll
    for (int d = 0; d < 8; ++d)
      O[((size_t)batch * NSEQ + m) * DMODEL + d * 16 + lane16] = (__bf16)(oacc[d][r] * inv);
  }
}

// ---------------------------------------------------------------------------
extern "C" void kernel_launch(void* const* d_in, const int* in_sizes, int n_in,
                              void* d_out, int out_size, void* d_ws, size_t ws_size,
                              hipStream_t stream) {
  const float* query = (const float*)d_in[0];
  const float* key_  = (const float*)d_in[1];
  const unsigned char* mask = (const unsigned char*)d_in[2];  // jnp bool_ = 1 byte
  const float* Wq = (const float*)d_in[3];  const float* bq = (const float*)d_in[4];
  const float* Wk = (const float*)d_in[5];  const float* bk = (const float*)d_in[6];
  const float* Wv = (const float*)d_in[7];  const float* bv = (const float*)d_in[8];
  const float* Wo = (const float*)d_in[9];  const float* bo = (const float*)d_in[10];
  float* out = (float*)d_out;

  const int rows = NB * NSEQ;                      // 32768
  const size_t tensor_bytes = (size_t)rows * DMODEL * sizeof(__bf16);  // 8 MB
  __bf16* qws = (__bf16*)d_ws;
  __bf16* kws = (__bf16*)((char*)d_ws + tensor_bytes);
  __bf16* vws = (__bf16*)((char*)d_ws + 2 * tensor_bytes);
  __bf16* aws = (__bf16*)((char*)d_ws + 3 * tensor_bytes);

  const int gemm_blocks = rows * 8 / 16 / 4;       // 4096 (4 tiles/block)

  gemm_d128<false, false><<<gemm_blocks, 128, 0, stream>>>(query, Wq, bq, qws, rows);
  gemm_d128<false, false><<<gemm_blocks, 128, 0, stream>>>(key_,  Wk, bk, kws, rows);
  gemm_d128<true,  false><<<gemm_blocks, 128, 0, stream>>>(kws,   Wv, bv, vws, rows);

  attn_flash<<<dim3(NSEQ / 16 / 4, NB), 128, 0, stream>>>(qws, kws, vws, mask, aws);

  gemm_d128<true, true><<<gemm_blocks, 128, 0, stream>>>(aws, Wo, bo, out, rows);
}
